// InvariantPolynomial_50560355008762
// MI455X (gfx1250) — matrix operand order repaired and checked by reading, not compiled, not run on hardware
//
#include <hip/hip_runtime.h>
#include <hip/hip_bf16.h>

#define N_NODES 50000
#define N_EDGES 400000
#define N_MOLS  2000
#define N_ATOM  23
#define N_BOND  7
#define MID1C   54   // compact mid1: 16 (l0+) + 18 (l1-) + 20 (l2+)
#define MID2    56

typedef __attribute__((ext_vector_type(16))) _Float16 v16h;
typedef __attribute__((ext_vector_type(8)))  float    v8f;

// ---------------------------------------------------------------------------
// Wigner-3j coefficient helpers (real basis, e3nn ordering: l=1 -> (y,z,x),
// l=2 -> (xy, yz, 3z^2-r^2, xz, x^2-y^2)). constexpr so fully-unrolled loops
// constant-fold and zero couplings disappear.
// ---------------------------------------------------------------------------
constexpr float pairAS(int j, int k, int j0, int k0, float v) {
    return (j == j0 && k == k0) ? v : ((j == k0 && k == j0) ? -v : 0.f);
}
// C(1,1,2): symmetric 1x1->2 coupling
constexpr float c112f(int i, int j, int k) {
    const float c = 0.31622777f;   // 1/sqrt(10)
    const float d = 0.18257419f;   // 1/sqrt(30)
    if (k == 0) return ((i == 2 && j == 0) || (i == 0 && j == 2)) ? c : 0.f;
    if (k == 1) return ((i == 0 && j == 1) || (i == 1 && j == 0)) ? c : 0.f;
    if (k == 2) return (i == j) ? (i == 1 ? 2.f * d : -d) : 0.f;
    if (k == 3) return ((i == 1 && j == 2) || (i == 2 && j == 1)) ? c : 0.f;
    return (i == j) ? (i == 2 ? c : (i == 0 ? -c : 0.f)) : 0.f;
}
// C(1,2,2): antisymmetric coupling = SO(3) generators on l=2 / sqrt(30)
constexpr float c122f(int i, int j, int k) {
    const float A = 0.18257419f;   // 1/sqrt(30)
    const float B = 0.31622777f;   // sqrt(3)/sqrt(30)
    const float C2 = 0.36514837f;  // 2/sqrt(30)
    if (i == 0) return pairAS(j, k, 0, 1, A) + pairAS(j, k, 2, 3, -B) + pairAS(j, k, 3, 4, -A);
    if (i == 1) return pairAS(j, k, 0, 4, C2) + pairAS(j, k, 1, 3, A);
    return pairAS(j, k, 0, 3, -A) + pairAS(j, k, 1, 2, -B) + pairAS(j, k, 1, 4, -A);
}
constexpr float c121f(int i, int j, int k) { return c112f(i, k, j); }
constexpr float c211f(int i, int j, int k) { return c112f(j, k, i); }
constexpr float c212f(int i, int j, int k) { return c122f(j, i, k); }
constexpr float c221f(int i, int j, int k) { return c122f(k, i, j); }
// C(2,2,2): symmetric 2x2->2 coupling (d-symbol pattern)
constexpr float dpat5(int j) {
    return (j == 0 || j == 4) ? 0.25197632f : (j == 2 ? -0.25197632f : -0.12598816f);
}
constexpr bool perm013(int i, int j, int k) {
    return (i != j) && (j != k) && (i != k) &&
           (i == 0 || i == 1 || i == 3) && (j == 0 || j == 1 || j == 3) &&
           (k == 0 || k == 1 || k == 3);
}
constexpr float c222f(int i, int j, int k) {
    float v = 0.f;
    if (i == 2 && j == k) v += dpat5(j);
    else if (j == 2 && i == k) v += dpat5(i);
    else if (k == 2 && i == j) v += dpat5(i);
    if (perm013(i, j, k)) v += 0.28f;
    if ((i == 4 && j == 1 && k == 1) || (j == 4 && i == 1 && k == 1) || (k == 4 && i == 1 && j == 1)) v -= 0.21f;
    if ((i == 4 && j == 3 && k == 3) || (j == 4 && i == 3 && k == 3) || (k == 4 && i == 3 && j == 3)) v += 0.21f;
    return v;
}

// ---------------------------------------------------------------------------
// K0: degree accumulation + K1: dis = 1/sqrt(deg)
// ---------------------------------------------------------------------------
__global__ __launch_bounds__(256) void k_deg(const int* __restrict__ edst, float* __restrict__ deg) {
    int e = blockIdx.x * 256 + threadIdx.x;
    if (e < N_EDGES) atomicAdd(&deg[edst[e]], 1.0f);
}
__global__ __launch_bounds__(256) void k_dis(const float* __restrict__ deg, float* __restrict__ dis) {
    int n = blockIdx.x * 256 + threadIdx.x;
    if (n < N_NODES) dis[n] = rsqrtf(deg[n]);
}

// ---------------------------------------------------------------------------
// K2: per-edge spherical harmonics * edge_attr[:,0]
// ---------------------------------------------------------------------------
__global__ __launch_bounds__(256) void k_eattr(const float* __restrict__ pos,
                                               const float* __restrict__ edge_attr,
                                               const int* __restrict__ esrc,
                                               const int* __restrict__ edst,
                                               float* __restrict__ eattr) {
    int e = blockIdx.x * 256 + threadIdx.x;
    if (e >= N_EDGES) return;
    int s = esrc[e], d = edst[e];
    float x = pos[s * 3 + 0] - pos[d * 3 + 0];
    float y = pos[s * 3 + 1] - pos[d * 3 + 1];
    float z = pos[s * 3 + 2] - pos[d * 3 + 2];
    float r2 = x * x + y * y + z * z;
    const float s3 = 1.73205081f, s5 = 2.23606798f, s15 = 3.87298335f;
    float ea0 = edge_attr[(size_t)e * N_BOND + 0];
    float sh[9] = {1.f, s3 * y, s3 * z, s3 * x, s15 * x * y, s15 * y * z,
                   0.5f * s5 * (3.f * z * z - r2), s15 * x * z, 0.5f * s15 * (x * x - y * y)};
#pragma unroll
    for (int j = 0; j < 9; ++j) eattr[(size_t)e * 9 + j] = ea0 * sh[j];
}

// ---------------------------------------------------------------------------
// K3: fused gather + linear(23x23)+bias+ReLU + TP1 (23->26 GEMM) + scatter.
// One wave per 16-edge tile; 4 WMMA f16 ops per wave. K padded 23->32.
// ---------------------------------------------------------------------------
__device__ inline _Float16 w1col(const float* __restrict__ w, int k, int col) {
    if (k >= N_ATOM) return (_Float16)0.f;
    float v = 0.f;
    if (col < 16)      v = w[k * 16 + col];            // l0 block (23x16)
    else if (col < 22) v = w[368 + k * 6 + (col - 16)]; // l1 block (23x6)
    else if (col < 26) v = w[506 + k * 4 + (col - 22)]; // l2 block (23x4)
    return (_Float16)v;
}

__global__ __launch_bounds__(128) void k_lin_tp1(const float* __restrict__ x,
                                                 const float* __restrict__ lin_w,
                                                 const float* __restrict__ lin_b,
                                                 const float* __restrict__ w_tp1,
                                                 const float* __restrict__ eattr,
                                                 const int* __restrict__ esrc,
                                                 const int* __restrict__ edst,
                                                 float* __restrict__ nf1) {
    __shared__ _Float16 s_first[4][16 * 32];
    __shared__ float    s_ea[4][16 * 9];
    __shared__ int      s_dst[4][16];

    const int wv = threadIdx.x >> 5;
    const int lane = threadIdx.x & 31;
    const int base = blockIdx.x * 64 + wv * 16;

    // stage eattr + dst for this tile
    for (int i = lane; i < 16 * 9; i += 32)
        s_ea[wv][i] = eattr[(size_t)(base + i / 9) * 9 + (i % 9)];
    if (lane < 16) s_dst[wv][lane] = edst[base + lane];

    const int m = lane & 15;      // row within tile
    const int kh = lane >> 4;     // K half-group
    const int row = esrc[base + m];

    // A fragment: ISA 16-bit A 16x32 layout: v<4 -> K = kh*8+2v, v>=4 -> +16
    v16h a{};
#pragma unroll
    for (int v = 0; v < 8; ++v) {
        const int kk = ((v >> 2) << 4) + kh * 8 + ((v & 3) << 1);
        a[2 * v]     = (kk < N_ATOM)     ? (_Float16)x[(size_t)row * N_ATOM + kk]     : (_Float16)0.f;
        a[2 * v + 1] = (kk + 1 < N_ATOM) ? (_Float16)x[(size_t)row * N_ATOM + kk + 1] : (_Float16)0.f;
    }
    // B fragments (32x16): same K order as A: v<4 -> K = kh*8+2v, v>=4 -> +16
    const int n = m, n2 = 16 + m;
    v16h b1a{}, b1b{};
#pragma unroll
    for (int v = 0; v < 8; ++v) {
        const int kk = ((v >> 2) << 4) + kh * 8 + ((v & 3) << 1);
        b1a[2 * v]     = (kk < N_ATOM)                 ? (_Float16)lin_w[n * N_ATOM + kk]      : (_Float16)0.f;
        b1a[2 * v + 1] = (kk + 1 < N_ATOM)             ? (_Float16)lin_w[n * N_ATOM + kk + 1]  : (_Float16)0.f;
        b1b[2 * v]     = (n2 < N_ATOM && kk < N_ATOM)     ? (_Float16)lin_w[n2 * N_ATOM + kk]     : (_Float16)0.f;
        b1b[2 * v + 1] = (n2 < N_ATOM && kk + 1 < N_ATOM) ? (_Float16)lin_w[n2 * N_ATOM + kk + 1] : (_Float16)0.f;
    }
    // C = broadcast bias over rows
    v8f c1a, c1b;
    const float ba = lin_b[n];
    const float bb = (n2 < N_ATOM) ? lin_b[n2] : 0.f;
#pragma unroll
    for (int r = 0; r < 8; ++r) { c1a[r] = ba; c1b[r] = bb; }

    v8f d1a = __builtin_amdgcn_wmma_f32_16x16x32_f16(false, a, false, b1a, (short)0, c1a, false, false);
    v8f d1b = __builtin_amdgcn_wmma_f32_16x16x32_f16(false, a, false, b1b, (short)0, c1b, false, false);

    __syncthreads();
    // ReLU -> LDS (f16, 16x32, zero-padded K)
#pragma unroll
    for (int r = 0; r < 8; ++r) {
        const int mr = r + kh * 8;
        float fa = d1a[r]; fa = fa > 0.f ? fa : 0.f;
        float fb = d1b[r]; fb = (n2 < N_ATOM && fb > 0.f) ? fb : 0.f;
        s_first[wv][mr * 32 + n]      = (_Float16)fa;
        s_first[wv][mr * 32 + 16 + n] = (_Float16)fb;
    }
    __syncthreads();

    // Second GEMM: first(16x23) @ Wcols(23x26) for TP1
    v16h a2{};
#pragma unroll
    for (int v = 0; v < 8; ++v) {
        const int kk = ((v >> 2) << 4) + kh * 8 + ((v & 3) << 1);
        a2[2 * v]     = s_first[wv][m * 32 + kk];
        a2[2 * v + 1] = s_first[wv][m * 32 + kk + 1];
    }
    v16h b2a{}, b2b{};
#pragma unroll
    for (int v = 0; v < 8; ++v) {
        const int kk = ((v >> 2) << 4) + kh * 8 + ((v & 3) << 1);
        b2a[2 * v]     = w1col(w_tp1, kk, n);
        b2a[2 * v + 1] = w1col(w_tp1, kk + 1, n);
        b2b[2 * v]     = w1col(w_tp1, kk, n2);
        b2b[2 * v + 1] = w1col(w_tp1, kk + 1, n2);
    }
    v8f z{};
    v8f d2a = __builtin_amdgcn_wmma_f32_16x16x32_f16(false, a2, false, b2a, (short)0, z, false, false);
    v8f d2b = __builtin_amdgcn_wmma_f32_16x16x32_f16(false, a2, false, b2b, (short)0, z, false, false);

    // TP1 scatter: every path coefficient is exactly 1/sqrt(23)
    const float C23 = 0.20851441f;
#pragma unroll
    for (int r = 0; r < 8; ++r) {
        const int mr = r + kh * 8;
        float* nfrow = nf1 + (size_t)s_dst[wv][mr] * MID1C;
        const float* ea = &s_ea[wv][mr * 9];
        atomicAdd(&nfrow[n], C23 * d2a[r] * ea[0]);          // l0 cols 0..15
        const float t = d2b[r];
        if (n < 6) {                                          // l1- cols 16..21
#pragma unroll
            for (int k = 0; k < 3; ++k)
                atomicAdd(&nfrow[16 + n * 3 + k], C23 * t * ea[1 + k]);
        } else if (n < 10) {                                  // l2+ cols 22..25
#pragma unroll
            for (int k = 0; k < 5; ++k)
                atomicAdd(&nfrow[34 + (n - 6) * 5 + k], C23 * t * ea[4 + k]);
        }
    }
}

// ---------------------------------------------------------------------------
// K4: scale node features by dis (generic)
// ---------------------------------------------------------------------------
__global__ __launch_bounds__(256) void k_scale(float* __restrict__ buf,
                                               const float* __restrict__ dis,
                                               int cols, int total) {
    int idx = blockIdx.x * 256 + threadIdx.x;
    if (idx < total) buf[idx] *= dis[idx / cols];
}

// ---------------------------------------------------------------------------
// K5: TP2 (mid1 x sh -> mid2), one thread per edge. Paths from zero chunks
// (i1 in {1,4}) are skipped (inputs identically zero); weight offsets follow
// the full reference path order.
// ---------------------------------------------------------------------------
__global__ __launch_bounds__(256) void k_tp2(const float* __restrict__ nf1,
                                             const float* __restrict__ eattr,
                                             const float* __restrict__ w_tp2,
                                             const int* __restrict__ esrc,
                                             const int* __restrict__ edst,
                                             float* __restrict__ nf2) {
    __shared__ float sw[588];
    for (int i = threadIdx.x; i < 588; i += 256) sw[i] = w_tp2[i];
    __syncthreads();
    const int e = blockIdx.x * 256 + threadIdx.x;
    if (e >= N_EDGES) return;

    const float* a = nf1 + (size_t)esrc[e] * MID1C;
    float A0[16], A2[6][3], A3[4][5], b[9];
#pragma unroll
    for (int u = 0; u < 16; ++u) A0[u] = a[u];
#pragma unroll
    for (int u = 0; u < 6; ++u)
#pragma unroll
        for (int i = 0; i < 3; ++i) A2[u][i] = a[16 + u * 3 + i];
#pragma unroll
    for (int u = 0; u < 4; ++u)
#pragma unroll
        for (int i = 0; i < 5; ++i) A3[u][i] = a[34 + u * 5 + i];
#pragma unroll
    for (int j = 0; j < 9; ++j) b[j] = eattr[(size_t)e * 9 + j];

    float o0[8] = {}, o1[9] = {}, o2[9] = {}, o3[10] = {}, o4[20] = {};
    const float PW0 = 0.19611614f;  // sqrt(1/26)
    const float PW1 = 0.33968311f;  // sqrt(3/26)
    const float PW2 = 0.28867513f;  // sqrt(3/36)
    const float PW3 = 0.35355339f;  // sqrt(5/40)
    const float PW4 = 0.45643546f;  // sqrt(5/24)
    const float IS3 = 0.57735027f, IS5 = 0.44721360f, IS6 = 0.40824829f;

    { // path (0,0,0) off 0, W 16x8
        const float* W = sw + 0;
#pragma unroll
        for (int w = 0; w < 8; ++w) {
            float s = 0.f;
#pragma unroll
            for (int u = 0; u < 16; ++u) s += A0[u] * W[u * 8 + w];
            o0[w] += PW0 * b[0] * s;
        }
    }
    { // path (0,1,2) off 128, W 16x3
        const float* W = sw + 128;
#pragma unroll
        for (int w = 0; w < 3; ++w) {
            float s = 0.f;
#pragma unroll
            for (int u = 0; u < 16; ++u) s += A0[u] * W[u * 3 + w];
#pragma unroll
            for (int k = 0; k < 3; ++k) o2[w * 3 + k] += PW2 * IS3 * b[1 + k] * s;
        }
    }
    { // path (0,2,3) off 176, W 16x2
        const float* W = sw + 176;
#pragma unroll
        for (int w = 0; w < 2; ++w) {
            float s = 0.f;
#pragma unroll
            for (int u = 0; u < 16; ++u) s += A0[u] * W[u * 2 + w];
#pragma unroll
            for (int k = 0; k < 5; ++k) o3[w * 5 + k] += PW3 * IS5 * b[4 + k] * s;
        }
    }
    // paths with i1=1 (zero input) occupy weights [208,298): skipped
    { // path (2,0,2) off 298, W 6x3
        const float* W = sw + 298;
#pragma unroll
        for (int w = 0; w < 3; ++w) {
            float s[3] = {};
#pragma unroll
            for (int u = 0; u < 6; ++u)
#pragma unroll
                for (int i = 0; i < 3; ++i) s[i] += A2[u][i] * W[u * 3 + w];
#pragma unroll
            for (int k = 0; k < 3; ++k) o2[w * 3 + k] += PW2 * IS3 * b[0] * s[k];
        }
    }
    { // path (2,1,0) off 316, W 6x8, C(1,1,0)=delta/sqrt3
        const float* W = sw + 316;
#pragma unroll
        for (int w = 0; w < 8; ++w) {
            float acc = 0.f;
#pragma unroll
            for (int i = 0; i < 3; ++i) {
                float s = 0.f;
#pragma unroll
                for (int u = 0; u < 6; ++u) s += A2[u][i] * W[u * 8 + w];
                acc += s * b[1 + i];
            }
            o0[w] += PW0 * IS3 * acc;
        }
    }
    { // path (2,1,1) off 364, W 6x3, C(1,1,1)=eps/sqrt6 (cross product)
        const float* W = sw + 364;
#pragma unroll
        for (int w = 0; w < 3; ++w) {
            float s0 = 0.f, s1 = 0.f, s2 = 0.f;
#pragma unroll
            for (int u = 0; u < 6; ++u) {
                float wv = W[u * 3 + w];
                s0 += A2[u][0] * wv; s1 += A2[u][1] * wv; s2 += A2[u][2] * wv;
            }
            o1[w * 3 + 0] += PW1 * IS6 * (s1 * b[3] - s2 * b[2]);
            o1[w * 3 + 1] += PW1 * IS6 * (s2 * b[1] - s0 * b[3]);
            o1[w * 3 + 2] += PW1 * IS6 * (s0 * b[2] - s1 * b[1]);
        }
    }
    { // path (2,1,3) off 382, W 6x2, C(1,1,2)
        const float* W = sw + 382;
#pragma unroll
        for (int w = 0; w < 2; ++w) {
            float s[3] = {};
#pragma unroll
            for (int u = 0; u < 6; ++u)
#pragma unroll
                for (int i = 0; i < 3; ++i) s[i] += A2[u][i] * W[u * 2 + w];
#pragma unroll
            for (int k = 0; k < 5; ++k) {
                float acc = 0.f;
#pragma unroll
                for (int i = 0; i < 3; ++i)
#pragma unroll
                    for (int j = 0; j < 3; ++j) acc += c112f(i, j, k) * s[i] * b[1 + j];
                o3[w * 5 + k] += PW3 * acc;
            }
        }
    }
    { // path (2,2,2) off 394, W 6x3, C(1,2,1)
        const float* W = sw + 394;
#pragma unroll
        for (int w = 0; w < 3; ++w) {
            float s[3] = {};
#pragma unroll
            for (int u = 0; u < 6; ++u)
#pragma unroll
                for (int i = 0; i < 3; ++i) s[i] += A2[u][i] * W[u * 3 + w];
#pragma unroll
            for (int k = 0; k < 3; ++k) {
                float acc = 0.f;
#pragma unroll
                for (int i = 0; i < 3; ++i)
#pragma unroll
                    for (int j = 0; j < 5; ++j) acc += c121f(i, j, k) * s[i] * b[4 + j];
                o2[w * 3 + k] += PW2 * acc;
            }
        }
    }
    { // path (2,2,4) off 412, W 6x4, C(1,2,2)
        const float* W = sw + 412;
#pragma unroll
        for (int w = 0; w < 4; ++w) {
            float s[3] = {};
#pragma unroll
            for (int u = 0; u < 6; ++u)
#pragma unroll
                for (int i = 0; i < 3; ++i) s[i] += A2[u][i] * W[u * 4 + w];
#pragma unroll
            for (int k = 0; k < 5; ++k) {
                float acc = 0.f;
#pragma unroll
                for (int i = 0; i < 3; ++i)
#pragma unroll
                    for (int j = 0; j < 5; ++j) acc += c122f(i, j, k) * s[i] * b[4 + j];
                o4[w * 5 + k] += PW4 * acc;
            }
        }
    }
    { // path (3,0,3) off 436, W 4x2
        const float* W = sw + 436;
#pragma unroll
        for (int w = 0; w < 2; ++w) {
            float s[5] = {};
#pragma unroll
            for (int u = 0; u < 4; ++u)
#pragma unroll
                for (int i = 0; i < 5; ++i) s[i] += A3[u][i] * W[u * 2 + w];
#pragma unroll
            for (int k = 0; k < 5; ++k) o3[w * 5 + k] += PW3 * IS5 * b[0] * s[k];
        }
    }
    { // path (3,1,2) off 444, W 4x3, C(2,1,1)
        const float* W = sw + 444;
#pragma unroll
        for (int w = 0; w < 3; ++w) {
            float s[5] = {};
#pragma unroll
            for (int u = 0; u < 4; ++u)
#pragma unroll
                for (int i = 0; i < 5; ++i) s[i] += A3[u][i] * W[u * 3 + w];
#pragma unroll
            for (int k = 0; k < 3; ++k) {
                float acc = 0.f;
#pragma unroll
                for (int i = 0; i < 5; ++i)
#pragma unroll
                    for (int j = 0; j < 3; ++j) acc += c211f(i, j, k) * s[i] * b[1 + j];
                o2[w * 3 + k] += PW2 * acc;
            }
        }
    }
    { // path (3,1,4) off 456, W 4x4, C(2,1,2)
        const float* W = sw + 456;
#pragma unroll
        for (int w = 0; w < 4; ++w) {
            float s[5] = {};
#pragma unroll
            for (int u = 0; u < 4; ++u)
#pragma unroll
                for (int i = 0; i < 5; ++i) s[i] += A3[u][i] * W[u * 4 + w];
#pragma unroll
            for (int k = 0; k < 5; ++k) {
                float acc = 0.f;
#pragma unroll
                for (int i = 0; i < 5; ++i)
#pragma unroll
                    for (int j = 0; j < 3; ++j) acc += c212f(i, j, k) * s[i] * b[1 + j];
                o4[w * 5 + k] += PW4 * acc;
            }
        }
    }
    { // path (3,2,0) off 472, W 4x8, C(2,2,0)=delta/sqrt5
        const float* W = sw + 472;
#pragma unroll
        for (int w = 0; w < 8; ++w) {
            float acc = 0.f;
#pragma unroll
            for (int i = 0; i < 5; ++i) {
                float s = 0.f;
#pragma unroll
                for (int u = 0; u < 4; ++u) s += A3[u][i] * W[u * 8 + w];
                acc += s * b[4 + i];
            }
            o0[w] += PW0 * IS5 * acc;
        }
    }
    { // path (3,2,1) off 504, W 4x3, C(2,2,1)
        const float* W = sw + 504;
#pragma unroll
        for (int w = 0; w < 3; ++w) {
            float s[5] = {};
#pragma unroll
            for (int u = 0; u < 4; ++u)
#pragma unroll
                for (int i = 0; i < 5; ++i) s[i] += A3[u][i] * W[u * 3 + w];
#pragma unroll
            for (int k = 0; k < 3; ++k) {
                float acc = 0.f;
#pragma unroll
                for (int i = 0; i < 5; ++i)
#pragma unroll
                    for (int j = 0; j < 5; ++j) acc += c221f(i, j, k) * s[i] * b[4 + j];
                o1[w * 3 + k] += PW1 * acc;
            }
        }
    }
    { // path (3,2,3) off 516, W 4x2, C(2,2,2)
        const float* W = sw + 516;
#pragma unroll
        for (int w = 0; w < 2; ++w) {
            float s[5] = {};
#pragma unroll
            for (int u = 0; u < 4; ++u)
#pragma unroll
                for (int i = 0; i < 5; ++i) s[i] += A3[u][i] * W[u * 2 + w];
#pragma unroll
            for (int k = 0; k < 5; ++k) {
                float acc = 0.f;
#pragma unroll
                for (int i = 0; i < 5; ++i)
#pragma unroll
                    for (int j = 0; j < 5; ++j) acc += c222f(i, j, k) * s[i] * b[4 + j];
                o3[w * 5 + k] += PW3 * acc;
            }
        }
    }
    // paths with i1=4 (zero input) occupy weights [524,588): skipped

    float* orow = nf2 + (size_t)edst[e] * MID2;
#pragma unroll
    for (int i = 0; i < 8; ++i)  atomicAdd(&orow[i], o0[i]);
#pragma unroll
    for (int i = 0; i < 9; ++i)  atomicAdd(&orow[8 + i], o1[i]);
#pragma unroll
    for (int i = 0; i < 9; ++i)  atomicAdd(&orow[17 + i], o2[i]);
#pragma unroll
    for (int i = 0; i < 10; ++i) atomicAdd(&orow[26 + i], o3[i]);
#pragma unroll
    for (int i = 0; i < 20; ++i) atomicAdd(&orow[36 + i], o4[i]);
}

// ---------------------------------------------------------------------------
// K6: TP3 (mid2 x sh -> scalar), one thread per edge
// ---------------------------------------------------------------------------
__global__ __launch_bounds__(256) void k_tp3(const float* __restrict__ nf2,
                                             const float* __restrict__ eattr,
                                             const float* __restrict__ w3,
                                             const int* __restrict__ esrc,
                                             const int* __restrict__ edst,
                                             float* __restrict__ nf3) {
    const int e = blockIdx.x * 256 + threadIdx.x;
    if (e >= N_EDGES) return;
    const float* a = nf2 + (size_t)esrc[e] * MID2;
    float b[9];
#pragma unroll
    for (int j = 0; j < 9; ++j) b[j] = eattr[(size_t)e * 9 + j];

    float acc = 0.f;
#pragma unroll
    for (int u = 0; u < 8; ++u) acc += a[u] * w3[u];  // path (0,0,0)
    acc *= b[0];
    float t1 = 0.f;                                   // path (2,1,0): l1- chunk @17
#pragma unroll
    for (int u = 0; u < 3; ++u) {
        float s = w3[8 + u];
#pragma unroll
        for (int i = 0; i < 3; ++i) t1 += a[17 + u * 3 + i] * b[1 + i] * s;
    }
    acc += 0.57735027f * t1;
    float t2 = 0.f;                                   // path (3,2,0): l2+ chunk @26
#pragma unroll
    for (int u = 0; u < 2; ++u) {
        float s = w3[11 + u];
#pragma unroll
        for (int i = 0; i < 5; ++i) t2 += a[26 + u * 5 + i] * b[4 + i] * s;
    }
    acc += 0.44721360f * t2;
    atomicAdd(&nf3[edst[e]], 0.27735010f * acc);      // pw = sqrt(1/13)
}

// ---------------------------------------------------------------------------
// K7/K8: molecule pooling
// ---------------------------------------------------------------------------
__global__ __launch_bounds__(256) void k_pool(const float* __restrict__ nf3,
                                              const float* __restrict__ dis,
                                              const int* __restrict__ batch,
                                              float* __restrict__ acc,
                                              float* __restrict__ apm) {
    int n = blockIdx.x * 256 + threadIdx.x;
    if (n >= N_NODES) return;
    int m = batch[n];
    atomicAdd(&acc[m], nf3[n] * dis[n]);
    atomicAdd(&apm[m], 1.0f);
}
__global__ __launch_bounds__(256) void k_final(const float* __restrict__ acc,
                                               const float* __restrict__ apm,
                                               float* __restrict__ out) {
    int m = blockIdx.x * 256 + threadIdx.x;
    if (m < N_MOLS) out[m] = acc[m] * rsqrtf(apm[m]);
}

// ---------------------------------------------------------------------------
extern "C" void kernel_launch(void* const* d_in, const int* in_sizes, int n_in,
                              void* d_out, int out_size, void* d_ws, size_t ws_size,
                              hipStream_t stream) {
    (void)in_sizes; (void)n_in; (void)out_size; (void)ws_size;
    const float* positions = (const float*)d_in[0];
    const float* x         = (const float*)d_in[1];
    const float* edge_attr = (const float*)d_in[2];
    const float* lin_w     = (const float*)d_in[3];
    const float* lin_b     = (const float*)d_in[4];
    const float* w_tp1     = (const float*)d_in[5];
    const float* w_tp2     = (const float*)d_in[6];
    const float* w_tp3     = (const float*)d_in[7];
    const int*   edge_src  = (const int*)d_in[8];
    const int*   edge_dst  = (const int*)d_in[9];
    const int*   batch     = (const int*)d_in[10];
    float* out = (float*)d_out;

    float* ws    = (float*)d_ws;
    float* deg   = ws;                                  // N
    float* dis   = deg + N_NODES;                       // N
    float* eattr = dis + N_NODES;                       // E*9
    float* nf1   = eattr + (size_t)N_EDGES * 9;         // N*54
    float* nf2   = nf1 + (size_t)N_NODES * MID1C;       // N*56
    float* nf3   = nf2 + (size_t)N_NODES * MID2;        // N
    float* acc   = nf3 + N_NODES;                       // M
    float* apm   = acc + N_MOLS;                        // M

    hipMemsetAsync(deg, 0, N_NODES * sizeof(float), stream);
    hipMemsetAsync(nf1, 0, (size_t)N_NODES * MID1C * sizeof(float), stream);
    hipMemsetAsync(nf2, 0, (size_t)N_NODES * MID2 * sizeof(float), stream);
    hipMemsetAsync(nf3, 0, N_NODES * sizeof(float), stream);
    hipMemsetAsync(acc, 0, N_MOLS * sizeof(float), stream);
    hipMemsetAsync(apm, 0, N_MOLS * sizeof(float), stream);

    const int EB = (N_EDGES + 255) / 256;
    const int NB = (N_NODES + 255) / 256;

    k_deg<<<EB, 256, 0, stream>>>(edge_dst, deg);
    k_dis<<<NB, 256, 0, stream>>>(deg, dis);
    k_eattr<<<EB, 256, 0, stream>>>(positions, edge_attr, edge_src, edge_dst, eattr);

    // fused linear+ReLU+TP1 with WMMA: 64 edges per 128-thread block
    k_lin_tp1<<<N_EDGES / 64, 128, 0, stream>>>(x, lin_w, lin_b, w_tp1, eattr,
                                                edge_src, edge_dst, nf1);
    {
        int tot = N_NODES * MID1C;
        k_scale<<<(tot + 255) / 256, 256, 0, stream>>>(nf1, dis, MID1C, tot);
    }
    k_tp2<<<EB, 256, 0, stream>>>(nf1, eattr, w_tp2, edge_src, edge_dst, nf2);
    {
        int tot = N_NODES * MID2;
        k_scale<<<(tot + 255) / 256, 256, 0, stream>>>(nf2, dis, MID2, tot);
    }
    k_tp3<<<EB, 256, 0, stream>>>(nf2, eattr, w_tp3, edge_src, edge_dst, nf3);
    k_pool<<<NB, 256, 0, stream>>>(nf3, dis, batch, acc, apm);
    k_final<<<(N_MOLS + 255) / 256, 256, 0, stream>>>(acc, apm, out);
}
